// Attention_5712306504559
// MI455X (gfx1250) — matrix operand order repaired and checked by reading, not compiled, not run on hardware
//
#include <hip/hip_runtime.h>
#include <hip/hip_bf16.h>

// ---------------------------------------------------------------------------
// MI455X (gfx1250) fused attention forward:
//   rmsnorm -> QKV GEMM (bf16 WMMA) -> head-L2-norm -> flash attention
//   (bf16 WMMA, online softmax, tanh softcap) -> output GEMM (bf16 WMMA, f32 out)
// All matrix math through v_wmma_f32_16x16x32_bf16 (wave32).
// ---------------------------------------------------------------------------

typedef __attribute__((ext_vector_type(16))) __bf16 v16bf;
typedef __attribute__((ext_vector_type(8)))  __bf16 v8bf;
typedef __attribute__((ext_vector_type(2)))  __bf16 v2bf;
typedef __attribute__((ext_vector_type(8)))  float  v8f;

#define DIM      1024
#define HEADS    16
#define DHEAD    64
#define BATCH    4
#define SEQ      2048
#define MTOT     (BATCH * SEQ)      // 8192 rows
#define NQKV     3072               // q(1024) | k(1024) | v(1024)

__device__ __forceinline__ v8f wmma_bf16(v16bf a, v16bf b, v8f c) {
  // 8 args: (neg_a, A, neg_b, B, c_mod, C, reuse_a, reuse_b)
  return __builtin_amdgcn_wmma_f32_16x16x32_bf16(false, a, false, b,
                                                 (short)0, c, false, false);
}

__device__ __forceinline__ float half_shfl_max(float v) {
  #pragma unroll
  for (int o = 1; o < 16; o <<= 1) v = fmaxf(v, __shfl_xor(v, o, 16));
  return v;
}
__device__ __forceinline__ float half_shfl_sum(float v) {
  #pragma unroll
  for (int o = 1; o < 16; o <<= 1) v += __shfl_xor(v, o, 16);
  return v;
}
// tanh via v_exp_f32 + v_rcp_f32 (avoids full-precision div expansion)
__device__ __forceinline__ float fast_tanh(float x) {
  x = fminf(fmaxf(x, -15.0f), 15.0f);
  float e = __expf(2.0f * x);
  return 1.0f - 2.0f * __builtin_amdgcn_rcpf(e + 1.0f);
}
// assemble v16bf A-fragment from two contiguous 16B chunks
__device__ __forceinline__ v16bf make_afrag(const __bf16* p) {
  v8bf c0 = *(const v8bf*)p;
  v8bf c1 = *(const v8bf*)(p + 8);
  v16bf a;
  #pragma unroll
  for (int e = 0; e < 8; ++e) { a[e] = c0[e]; a[e + 8] = c1[e]; }
  return a;
}

// ---------------------------------------------------------------------------
// Weight prep: fp32 [K][N] -> bf16 transposed [N][K] so WMMA B-fragments are
// contiguous 32B loads. Wq and Wkv are fused into one [3072][1024] buffer.
// ---------------------------------------------------------------------------
__global__ __launch_bounds__(256) void prep_wqkv_kernel(
    const float* __restrict__ Wq, const float* __restrict__ Wkv,
    __bf16* __restrict__ Wt) {
  int idx = blockIdx.x * 256 + threadIdx.x;          // over 3072*1024
  if (idx >= NQKV * DIM) return;
  int n = idx >> 10, k = idx & 1023;
  float v = (n < 1024) ? Wq[(size_t)k * 1024 + n]
                       : Wkv[(size_t)k * 2048 + (n - 1024)];
  Wt[idx] = (__bf16)v;
}

__global__ __launch_bounds__(256) void prep_wo_kernel(
    const float* __restrict__ Wo, __bf16* __restrict__ Wt) {
  int idx = blockIdx.x * 256 + threadIdx.x;          // over 1024*1024
  if (idx >= DIM * DIM) return;
  int n = idx >> 10, k = idx & 1023;
  Wt[idx] = (__bf16)Wo[(size_t)k * 1024 + n];
}

// ---------------------------------------------------------------------------
// RMSNorm: one 256-thread block per token row; output bf16.
// ---------------------------------------------------------------------------
__global__ __launch_bounds__(256) void rmsnorm_kernel(
    const float* __restrict__ tokens, const float* __restrict__ w,
    __bf16* __restrict__ Xb) {
  int m = blockIdx.x;
  const float* x = tokens + (size_t)m * DIM;
  float4 v = *(const float4*)(x + threadIdx.x * 4);
  float ss = v.x * v.x + v.y * v.y + v.z * v.z + v.w * v.w;
  #pragma unroll
  for (int o = 1; o < 32; o <<= 1) ss += __shfl_xor(ss, o, 32);
  __shared__ float red[8];
  int wave = threadIdx.x >> 5, lane = threadIdx.x & 31;
  if (lane == 0) red[wave] = ss;
  __syncthreads();
  float tot = red[0] + red[1] + red[2] + red[3] +
              red[4] + red[5] + red[6] + red[7];
  float inv = rsqrtf(tot * (1.0f / DIM) + 1.1920929e-07f);
  const float* wp = w + threadIdx.x * 4;
  __bf16* out = Xb + (size_t)m * DIM + threadIdx.x * 4;
  out[0] = (__bf16)(v.x * inv * wp[0]);
  out[1] = (__bf16)(v.y * inv * wp[1]);
  out[2] = (__bf16)(v.z * inv * wp[2]);
  out[3] = (__bf16)(v.w * inv * wp[3]);
}

// ---------------------------------------------------------------------------
// WMMA GEMM core, 32(M) x 64(N) per wave: C += A[M,K]bf16 x Bt[N,K]bf16.
// A-fragment: lane(lo,hi): a[e] = A[m0+lo][k0 + 16*hi + e]
//   -> two contiguous 16B loads at k0+16*hi and k0+16*hi+8.
// B-fragment: lane(lo,hi): b[e] = Bt[n0+lo][k0 + 16*hi + e]
//   -> one contiguous 32B load.
// 8 wmma per 12 b128 loads; pointer-increment induction; unconditional
// speculative prefetch (dropped on fault) -> no per-iteration branch.
// ---------------------------------------------------------------------------
__device__ __forceinline__ void gemm_core(
    const __bf16* __restrict__ A, const __bf16* __restrict__ Bt,
    int K, int m0, int n0, int lo, int hi, v8f (&c)[2][4]) {
  const __bf16* ap0 = A + (size_t)(m0 + lo) * K + 16 * hi;
  const __bf16* ap1 = A + (size_t)(m0 + 16 + lo) * K + 16 * hi;
  const __bf16* bp0 = Bt + (size_t)(n0 + lo) * K + 16 * hi;
  const __bf16* bp1 = Bt + (size_t)(n0 + 16 + lo) * K + 16 * hi;
  const __bf16* bp2 = Bt + (size_t)(n0 + 32 + lo) * K + 16 * hi;
  const __bf16* bp3 = Bt + (size_t)(n0 + 48 + lo) * K + 16 * hi;
  for (int k = 0; k < K; k += 32) {
    __builtin_prefetch(ap0 + 64, 0, 3);   // gfx1250 global_prefetch_b8
    __builtin_prefetch(bp0 + 64, 0, 3);
    __builtin_prefetch(bp2 + 64, 0, 3);
    v16bf a0 = make_afrag(ap0);
    v16bf a1 = make_afrag(ap1);
    v16bf b0 = *(const v16bf*)bp0;
    v16bf b1 = *(const v16bf*)bp1;
    v16bf b2 = *(const v16bf*)bp2;
    v16bf b3 = *(const v16bf*)bp3;
    c[0][0] = wmma_bf16(a0, b0, c[0][0]);
    c[0][1] = wmma_bf16(a0, b1, c[0][1]);
    c[0][2] = wmma_bf16(a0, b2, c[0][2]);
    c[0][3] = wmma_bf16(a0, b3, c[0][3]);
    c[1][0] = wmma_bf16(a1, b0, c[1][0]);
    c[1][1] = wmma_bf16(a1, b1, c[1][1]);
    c[1][2] = wmma_bf16(a1, b2, c[1][2]);
    c[1][3] = wmma_bf16(a1, b3, c[1][3]);
    ap0 += 32; ap1 += 32;
    bp0 += 32; bp1 += 32; bp2 += 32; bp3 += 32;
  }
}

// ---------------------------------------------------------------------------
// QKV GEMM: X[8192,1024] x Wt[3072,1024]^T. Epilogue scatters:
//   Q,K -> [b,h,n,64] bf16 ;  V -> transposed [b,h,64,n] bf16.
// Block = 256 thr = 8 waves (4M x 2N) covering 128(M) x 128(N).
// ---------------------------------------------------------------------------
__global__ __launch_bounds__(256) void gemm_qkv_kernel(
    const __bf16* __restrict__ X, const __bf16* __restrict__ Wt,
    __bf16* __restrict__ Qb, __bf16* __restrict__ Kb, __bf16* __restrict__ Vt) {
  int lane = threadIdx.x & 31, wave = threadIdx.x >> 5;
  int lo = lane & 15, hi = lane >> 4;
  int m0 = blockIdx.y * 128 + (wave & 3) * 32;
  int n0 = blockIdx.x * 128 + (wave >> 2) * 64;
  v8f c[2][4] = {};
  gemm_core(X, Wt, DIM, m0, n0, lo, hi, c);
  #pragma unroll
  for (int mt = 0; mt < 2; ++mt)
    #pragma unroll
    for (int nt = 0; nt < 4; ++nt)
      #pragma unroll
      for (int r = 0; r < 8; ++r) {
        int m = m0 + mt * 16 + r + 8 * hi;          // token row
        int cc = n0 + nt * 16 + lo;                 // fused output col
        float v = c[mt][nt][r];
        int which = cc >> 10;                       // 0=q 1=k 2=v
        int c1 = cc & 1023;
        int h = c1 >> 6, d = c1 & 63;
        int bi = m >> 11, ni = m & 2047;
        if (which == 0)
          Qb[(((size_t)bi * HEADS + h) * SEQ + ni) * DHEAD + d] = (__bf16)v;
        else if (which == 1)
          Kb[(((size_t)bi * HEADS + h) * SEQ + ni) * DHEAD + d] = (__bf16)v;
        else
          Vt[(((size_t)bi * HEADS + h) * DHEAD + d) * SEQ + ni] = (__bf16)v;
      }
}

// ---------------------------------------------------------------------------
// Per-head L2 norm: one wave32 per (b,h,n) row of 64; scale (gamma+1)*8.
// ---------------------------------------------------------------------------
__global__ __launch_bounds__(256) void headnorm_kernel(
    __bf16* __restrict__ T, const float* __restrict__ gamma) {
  int gid = blockIdx.x * 256 + threadIdx.x;
  int row = gid >> 5;
  if (row >= BATCH * HEADS * SEQ) return;
  int lane = threadIdx.x & 31;
  __bf16* rp = T + (size_t)row * DHEAD;
  v2bf x = *(const v2bf*)(rp + 2 * lane);
  float x0 = (float)x[0], x1 = (float)x[1];
  float ss = x0 * x0 + x1 * x1;
  #pragma unroll
  for (int o = 1; o < 32; o <<= 1) ss += __shfl_xor(ss, o, 32);
  float inv = __builtin_amdgcn_rcpf(fmaxf(sqrtf(ss), 1e-12f));
  int h = (row >> 11) & (HEADS - 1);
  float g0 = (gamma[h * DHEAD + 2 * lane]     + 1.0f) * 8.0f;  // sqrt(64)=8
  float g1 = (gamma[h * DHEAD + 2 * lane + 1] + 1.0f) * 8.0f;
  x[0] = (__bf16)(x0 * inv * g0);
  x[1] = (__bf16)(x1 * inv * g1);
  *(v2bf*)(rp + 2 * lane) = x;
}

// ---------------------------------------------------------------------------
// Flash attention: grid (16 qtile-groups, 64 b*h), 8 waves/block, one wave
// owns a 16-query tile and streams 32-key chunks with online softmax.
// Softcap: logit = 6.25 * tanh(0.02 * s)  (== tanh(s/50)*50/sqrt(64)).
// Scores->P(A-frag) transposed through a private 1KB/wave LDS tile.
// ---------------------------------------------------------------------------
__global__ __launch_bounds__(256) void attn_kernel(
    const __bf16* __restrict__ Q, const __bf16* __restrict__ K,
    const __bf16* __restrict__ Vt, __bf16* __restrict__ Ob) {
  __shared__ __attribute__((aligned(64))) __bf16 lds_p[8][16][32];
  int lane = threadIdx.x & 31, wave = threadIdx.x >> 5;
  int lo = lane & 15, hi = lane >> 4;
  int bh = blockIdx.y;                    // b*16 + h
  int q0 = blockIdx.x * 128 + wave * 16;
  const __bf16* Qp = Q  + (size_t)bh * SEQ * DHEAD;
  const __bf16* Kp = K  + (size_t)bh * SEQ * DHEAD;
  const __bf16* Vp = Vt + (size_t)bh * DHEAD * SEQ;

  // Q A-fragments for d=[0,32) and [32,64)
  v16bf aq0 = make_afrag(Qp + (size_t)(q0 + lo) * DHEAD + 16 * hi);
  v16bf aq1 = make_afrag(Qp + (size_t)(q0 + lo) * DHEAD + 32 + 16 * hi);

  // K fragment pointers: [subtile t2 0..1][d-half t 0..1], bump 32 rows/iter
  const __bf16* kp00 = Kp + (size_t)lo * DHEAD + 16 * hi;
  const __bf16* kp01 = kp00 + 32;
  const __bf16* kp10 = kp00 + 16 * DHEAD;
  const __bf16* kp11 = kp10 + 32;
  // V fragment pointers (transposed layout), bump 32 cols/iter
  const __bf16* vp0 = Vp + (size_t)lo * SEQ + 16 * hi;
  const __bf16* vp1 = vp0 + 16 * SEQ;
  const __bf16* vp2 = vp0 + 32 * SEQ;
  const __bf16* vp3 = vp0 + 48 * SEQ;

  v8f o[4] = {};
  float mrow[8], lrow[8];
  #pragma unroll
  for (int r = 0; r < 8; ++r) { mrow[r] = -1e30f; lrow[r] = 0.0f; }

  for (int j0 = 0; j0 < SEQ; j0 += 32) {
    __builtin_prefetch(kp00 + 32 * DHEAD, 0, 3);   // speculative, branch-free
    __builtin_prefetch(vp0 + 32, 0, 3);
    // --- scores for two 16-key subtiles ---
    v8f s0v = {}, s1v = {};
    {
      v16bf bk00 = *(const v16bf*)kp00;
      v16bf bk01 = *(const v16bf*)kp01;
      v16bf bk10 = *(const v16bf*)kp10;
      v16bf bk11 = *(const v16bf*)kp11;
      s0v = wmma_bf16(aq0, bk00, s0v);
      s0v = wmma_bf16(aq1, bk01, s0v);
      s1v = wmma_bf16(aq0, bk10, s1v);
      s1v = wmma_bf16(aq1, bk11, s1v);
    }
    // --- softcap + online softmax (row = r + 8*hi, cols = 16 lanes lo) ---
    #pragma unroll
    for (int r = 0; r < 8; ++r) {
      float s0 = 6.25f * fast_tanh(0.02f * s0v[r]);
      float s1 = 6.25f * fast_tanh(0.02f * s1v[r]);
      float mx = half_shfl_max(fmaxf(s0, s1));
      float mnew = fmaxf(mrow[r], mx);
      float corr = __expf(mrow[r] - mnew);
      float p0 = __expf(s0 - mnew);
      float p1 = __expf(s1 - mnew);
      lrow[r] = lrow[r] * corr + half_shfl_sum(p0 + p1);
      mrow[r] = mnew;
      #pragma unroll
      for (int f = 0; f < 4; ++f) o[f][r] *= corr;
      lds_p[wave][r + 8 * hi][lo]      = (__bf16)p0;
      lds_p[wave][r + 8 * hi][16 + lo] = (__bf16)p1;
    }
    // --- P as A-fragment (16 q rows x 32 keys) via LDS transpose bounce ---
    // (LDS ops are in-order within a wave; only this wave touches its tile)
    v16bf ap = make_afrag(&lds_p[wave][lo][0] + 16 * hi);
    // --- O += P @ V, B-fragments straight from transposed V (32B/lane) ---
    v16bf bv0 = *(const v16bf*)vp0;
    v16bf bv1 = *(const v16bf*)vp1;
    v16bf bv2 = *(const v16bf*)vp2;
    v16bf bv3 = *(const v16bf*)vp3;
    o[0] = wmma_bf16(ap, bv0, o[0]);
    o[1] = wmma_bf16(ap, bv1, o[1]);
    o[2] = wmma_bf16(ap, bv2, o[2]);
    o[3] = wmma_bf16(ap, bv3, o[3]);
    kp00 += 32 * DHEAD; kp01 += 32 * DHEAD;
    kp10 += 32 * DHEAD; kp11 += 32 * DHEAD;
    vp0 += 32; vp1 += 32; vp2 += 32; vp3 += 32;
  }
  // --- finalize: divide by row sums; store [b, n, h*64+d] bf16 ---
  int bq = bh >> 4, hq = bh & (HEADS - 1);
  #pragma unroll
  for (int r = 0; r < 8; ++r) {
    float rl = __builtin_amdgcn_rcpf(lrow[r]);
    int row = q0 + r + 8 * hi;
    #pragma unroll
    for (int f = 0; f < 4; ++f) {
      float val = o[f][r] * rl;
      Ob[((size_t)bq * SEQ + row) * DIM + hq * DHEAD + 16 * f + lo] = (__bf16)val;
    }
  }
}

// ---------------------------------------------------------------------------
// Output GEMM: O[8192,1024]bf16 x WoT[1024,1024]bf16 -> f32 d_out.
// ---------------------------------------------------------------------------
__global__ __launch_bounds__(256) void gemm_out_kernel(
    const __bf16* __restrict__ A, const __bf16* __restrict__ Wt,
    float* __restrict__ out) {
  int lane = threadIdx.x & 31, wave = threadIdx.x >> 5;
  int lo = lane & 15, hi = lane >> 4;
  int m0 = blockIdx.y * 128 + (wave & 3) * 32;
  int n0 = blockIdx.x * 128 + (wave >> 2) * 64;
  v8f c[2][4] = {};
  gemm_core(A, Wt, DIM, m0, n0, lo, hi, c);
  #pragma unroll
  for (int mt = 0; mt < 2; ++mt)
    #pragma unroll
    for (int nt = 0; nt < 4; ++nt)
      #pragma unroll
      for (int r = 0; r < 8; ++r) {
        int m = m0 + mt * 16 + r + 8 * hi;
        int cc = n0 + nt * 16 + lo;
        out[(size_t)m * DIM + cc] = c[mt][nt][r];
      }
}

// ---------------------------------------------------------------------------
// Launch. Workspace layout (bytes from d_ws, all 16B+ aligned):
//   Xb 16MB | WqkvT 6MB | WoT 2MB | Qb 16MB | Kb 16MB | Vt 16MB | Ob 16MB
// Total 88MB.
// ---------------------------------------------------------------------------
extern "C" void kernel_launch(void* const* d_in, const int* in_sizes, int n_in,
                              void* d_out, int out_size, void* d_ws, size_t ws_size,
                              hipStream_t stream) {
  (void)in_sizes; (void)n_in; (void)out_size; (void)ws_size;
  const float* tokens  = (const float*)d_in[0];
  const float* norm_w  = (const float*)d_in[1];
  const float* Wq      = (const float*)d_in[2];
  const float* Wkv     = (const float*)d_in[3];
  const float* Wo      = (const float*)d_in[4];
  const float* q_gamma = (const float*)d_in[5];
  const float* k_gamma = (const float*)d_in[6];
  float* out = (float*)d_out;

  char* ws = (char*)d_ws;
  __bf16* Xb    = (__bf16*)(ws);
  __bf16* WqkvT = (__bf16*)(ws + (size_t)16 * 1024 * 1024);
  __bf16* WoT   = (__bf16*)(ws + (size_t)22 * 1024 * 1024);
  __bf16* Qb    = (__bf16*)(ws + (size_t)24 * 1024 * 1024);
  __bf16* Kb    = (__bf16*)(ws + (size_t)40 * 1024 * 1024);
  __bf16* Vt    = (__bf16*)(ws + (size_t)56 * 1024 * 1024);
  __bf16* Ob    = (__bf16*)(ws + (size_t)72 * 1024 * 1024);

  prep_wqkv_kernel<<<(NQKV * DIM + 255) / 256, 256, 0, stream>>>(Wq, Wkv, WqkvT);
  prep_wo_kernel<<<(DIM * DIM + 255) / 256, 256, 0, stream>>>(Wo, WoT);
  rmsnorm_kernel<<<MTOT, 256, 0, stream>>>(tokens, norm_w, Xb);
  gemm_qkv_kernel<<<dim3(NQKV / 128, MTOT / 128), 256, 0, stream>>>(Xb, WqkvT, Qb, Kb, Vt);
  headnorm_kernel<<<(BATCH * HEADS * SEQ * 32 + 255) / 256, 256, 0, stream>>>(Qb, q_gamma);
  headnorm_kernel<<<(BATCH * HEADS * SEQ * 32 + 255) / 256, 256, 0, stream>>>(Kb, k_gamma);
  attn_kernel<<<dim3(SEQ / 128, BATCH * HEADS), 256, 0, stream>>>(Qb, Kb, Vt, Ob);
  gemm_out_kernel<<<dim3(DIM / 128, MTOT / 128), 256, 0, stream>>>(Ob, WoT, out);
}
